// EditDecoderRNN_11106785427749
// MI455X (gfx1250) — compile-verified, matrix operand order checked by reading, not run on hardware
//
#include <hip/hip_runtime.h>
#include <hip/hip_bf16.h>
#include <math.h>
#include <stdint.h>

// ---------------------------------------------------------------------------
// EditDecoderRNN for MI455X (gfx1250, wave32, WMMA).
// bf16 inputs + f32 accumulation via v_wmma_f32_16x16x32_bf16 for every GEMM.
// Big GEMMs use double-buffered LDS tiles staged with
// global_load_async_to_lds_b128 (ASYNCcnt) for load/compute overlap.
// Persistent-workgroup LSTM keeps hidden state in LDS as a WMMA A-tile.
// ---------------------------------------------------------------------------

typedef __bf16 bf16;
typedef __attribute__((ext_vector_type(16))) __bf16 v16bf;
typedef __attribute__((ext_vector_type(8)))  __bf16 v8bf;
typedef __attribute__((ext_vector_type(8)))  float  v8f;

#define DEV __device__ __forceinline__

namespace {
constexpr int KEEP_ID = 1, DEL_ID = 2, PAD_ID = 0, STOP_ID = 102, ANNO_ID = 103;
constexpr int Vv = 21128, Ee = 768, Hh = 768, Bb = 8, Tt = 256, Ss = 257;
constexpr int H4 = 4 * Hh;
}

// Per-lane WMMA operand load: for 16-bit A/B tiles the lane's 16 elements are
// K = hi*8 + [0..7] and K = 16 + hi*8 + [0..7]  ->  two contiguous 16B runs.
DEV v16bf ld_k32(const bf16* row, int kc, int hi) {
  const bf16* p = row + kc + hi * 8;
  v8bf lo = *(const v8bf*)p;
  v8bf hv = *(const v8bf*)(p + 16);
  v16bf r;
#pragma unroll
  for (int i = 0; i < 8; ++i) { r[i] = lo[i]; r[i + 8] = hv[i]; }
  return r;
}

DEV v8f wmma_bf16(v16bf a, v16bf b, v8f c) {
  return __builtin_amdgcn_wmma_f32_16x16x32_bf16(false, a, false, b, (short)0, c,
                                                 false, false);
}

// CDNA5 async global->LDS copy (16B per lane), tracked by ASYNCcnt.
DEV void async_b128(uint32_t lds_off, const bf16* gsrc) {
  asm volatile("global_load_async_to_lds_b128 %0, %1, off"
               :: "v"(lds_off), "v"(gsrc)
               : "memory");
}
DEV void wait_async0() { asm volatile("s_wait_asynccnt 0x0" ::: "memory"); }

// ---------------------------------------------------------------------------
// Block-tiled NT GEMM with async-LDS double buffering.
//   C[M,N] = act( A[M,K] * W[N,K]^T + bias[N] )
// Block = 256 threads (8 waves) computing a 64x128 C tile; wave (mi,nj) owns
// a 16x64 strip. K chunk = 32. LDS rows padded to 40 bf16 (80B: 16B-aligned
// b128 stores, bank-stride 20 -> conflict-free row reads).
// ---------------------------------------------------------------------------
__global__ __launch_bounds__(256) void gemm_nt_bf16_async(
    float* __restrict__ C, bf16* __restrict__ Cbf,
    const bf16* __restrict__ A, const bf16* __restrict__ W,
    const float* __restrict__ bias, int M, int N, int K, int ldc, int act) {
  __shared__ bf16 As[2][64][40];    // 10 KB
  __shared__ bf16 Bs[2][128][40];   // 20 KB

  const int m0 = blockIdx.y * 64, n0 = blockIdx.x * 128;
  const int tid = threadIdx.x, wave = tid >> 5, lane = tid & 31;
  const int mi = wave & 3, nj = wave >> 2;
  const int n = lane & 15, hi = lane >> 4;

  auto stage = [&](int buf, int kc) {
    {  // A tile: 64 rows x 32 bf16 = 256 lanes x 16B
      const int r = tid >> 2, seg = tid & 3;
      int gr = m0 + r;
      if (gr >= M) gr = M - 1;
      async_b128((uint32_t)(uintptr_t)&As[buf][r][seg * 8],
                 A + (size_t)gr * K + kc + seg * 8);
    }
#pragma unroll
    for (int c = 0; c < 2; ++c) {  // B tile: 128 rows x 32 bf16 = 512 x 16B
      const int chunk = tid + c * 256;
      const int r = chunk >> 2, seg = chunk & 3;
      int gr = n0 + r;
      if (gr >= N) gr = N - 1;
      async_b128((uint32_t)(uintptr_t)&Bs[buf][r][seg * 8],
                 W + (size_t)gr * K + kc + seg * 8);
    }
  };

  stage(0, 0);
  wait_async0();
  __syncthreads();

  v8f acc0 = {}, acc1 = {}, acc2 = {}, acc3 = {};
  int cur = 0;
  for (int kc = 0; kc < K; kc += 32, cur ^= 1) {
    if (kc + 32 < K) stage(cur ^ 1, kc + 32);   // overlap next stage with compute
    v16bf av = ld_k32(&As[cur][mi * 16 + (lane & 15)][0], 0, hi);
    acc0 = wmma_bf16(av, ld_k32(&Bs[cur][nj * 64 +  0 + n][0], 0, hi), acc0);
    acc1 = wmma_bf16(av, ld_k32(&Bs[cur][nj * 64 + 16 + n][0], 0, hi), acc1);
    acc2 = wmma_bf16(av, ld_k32(&Bs[cur][nj * 64 + 32 + n][0], 0, hi), acc2);
    acc3 = wmma_bf16(av, ld_k32(&Bs[cur][nj * 64 + 48 + n][0], 0, hi), acc3);
    wait_async0();
    __syncthreads();            // next buffer ready; all reads of cur complete
  }

  v8f* accs[4] = {&acc0, &acc1, &acc2, &acc3};
#pragma unroll
  for (int j = 0; j < 4; ++j) {
    const int cn = n0 + nj * 64 + j * 16 + n;
    if (cn >= N) continue;
    const float bb = bias ? bias[cn] : 0.0f;
#pragma unroll
    for (int r = 0; r < 8; ++r) {
      const int cm = m0 + mi * 16 + r + hi * 8;
      if (cm < M) {
        float v = (*accs[j])[r] + bb;
        if (act == 1) v = tanhf(v);
        C[(size_t)cm * ldc + cn] = v;
        if (Cbf) Cbf[(size_t)cm * ldc + cn] = (bf16)v;
      }
    }
  }
}

// ---------------------------------------------------------------------------
// Direct-load NT GEMM (register double-buffered), batched via gridDim.y.
// Used for the per-batch attention scores GEMM.
// ---------------------------------------------------------------------------
__global__ __launch_bounds__(256) void gemm_nt_bf16(
    float* __restrict__ C, bf16* __restrict__ Cbf,
    const bf16* __restrict__ A, const bf16* __restrict__ W,
    const float* __restrict__ bias, int M, int N, int K, int ldc, int act,
    long long sA, long long sW, long long sC) {
  const int batch = blockIdx.y;
  A += (size_t)batch * sA;
  W += (size_t)batch * sW;
  C += (size_t)batch * sC;

  const int gw = (blockIdx.x * blockDim.x + threadIdx.x) >> 5;
  const int lane = threadIdx.x & 31;
  const int mt = (M + 15) >> 4, ng = (N + 63) >> 6;
  if (gw >= mt * ng) return;
  const int mTile = gw / ng, nGrp = gw % ng;
  const int m0 = mTile * 16, n0 = nGrp * 64;
  const int n = lane & 15, hi = lane >> 4;

  int ra = m0 + (lane & 15);
  if (ra >= M) ra = M - 1;
  const bf16* arow = A + (size_t)ra * K;

  const bf16* wrow[4];
#pragma unroll
  for (int j = 0; j < 4; ++j) {
    int rb = n0 + j * 16 + n;
    if (rb >= N) rb = N - 1;
    wrow[j] = W + (size_t)rb * K;
  }

  v8f acc0 = {}, acc1 = {}, acc2 = {}, acc3 = {};
  // register double-buffer: fetch chunk k+1 while computing chunk k
  v16bf a_c = ld_k32(arow, 0, hi);
  v16bf b0 = ld_k32(wrow[0], 0, hi), b1 = ld_k32(wrow[1], 0, hi);
  v16bf b2 = ld_k32(wrow[2], 0, hi), b3 = ld_k32(wrow[3], 0, hi);
  for (int kc = 0; kc < K; kc += 32) {
    v16bf a_n, c0, c1, c2, c3;
    const int kn = kc + 32;
    if (kn < K) {
      __builtin_prefetch(arow + kn + 32, 0, 1);
      a_n = ld_k32(arow, kn, hi);
      c0 = ld_k32(wrow[0], kn, hi);
      c1 = ld_k32(wrow[1], kn, hi);
      c2 = ld_k32(wrow[2], kn, hi);
      c3 = ld_k32(wrow[3], kn, hi);
    }
    acc0 = wmma_bf16(a_c, b0, acc0);
    acc1 = wmma_bf16(a_c, b1, acc1);
    acc2 = wmma_bf16(a_c, b2, acc2);
    acc3 = wmma_bf16(a_c, b3, acc3);
    if (kn < K) { a_c = a_n; b0 = c0; b1 = c1; b2 = c2; b3 = c3; }
  }

  v8f* accs[4] = {&acc0, &acc1, &acc2, &acc3};
#pragma unroll
  for (int j = 0; j < 4; ++j) {
    const int cn = n0 + j * 16 + n;
    if (cn >= N) continue;
    const float bb = bias ? bias[cn] : 0.0f;
#pragma unroll
    for (int r = 0; r < 8; ++r) {
      const int cm = m0 + r + hi * 8;
      if (cm < M) {
        float v = (*accs[j])[r] + bb;
        if (act == 1) v = tanhf(v);
        C[(size_t)cm * ldc + cn] = v;
        if (Cbf) Cbf[(size_t)batch * sC + (size_t)cm * ldc + cn] = (bf16)v;
      }
    }
  }
}

// ---------------------------------------------------------------------------
// Persistent single-workgroup LSTM. 512 threads = 16 waves; wave w owns h-cols
// [48w, 48w+48). Hidden state h_t lives in LDS as a 16x768 bf16 WMMA A-tile
// (rows 8..15 zero). Cell state stays in registers. Gx = x@Wih^T precomputed.
// ---------------------------------------------------------------------------
__global__ __launch_bounds__(512) void lstm_seq_kernel(
    float* __restrict__ hs, bf16* __restrict__ hs_bf,
    float* __restrict__ hT, float* __restrict__ cT,
    const float* __restrict__ Gx, const bf16* __restrict__ Whh,
    const float* __restrict__ bvec, const float* __restrict__ h0,
    const float* __restrict__ c0, int steps) {
  __shared__ bf16 Hbf[16][Hh];  // 24 KB

  const int tid = threadIdx.x;
  const int wave = tid >> 5, lane = tid & 31;
  const int n = lane & 15, hi = lane >> 4;

  for (int i = tid; i < 16 * Hh; i += 512) {
    const int m = i / Hh, k = i % Hh;
    Hbf[m][k] = (m < Bb) ? (bf16)h0[m * Hh + k] : (bf16)0.0f;
  }

  float cst[3][8], h2l[3][8];
#pragma unroll
  for (int j = 0; j < 3; ++j)
#pragma unroll
    for (int r = 0; r < 8; ++r) {
      cst[j][r] = (hi == 0) ? c0[r * Hh + wave * 48 + j * 16 + n] : 0.0f;
      h2l[j][r] = 0.0f;
    }
  __syncthreads();

  for (int t = 0; t < steps; ++t) {
#pragma unroll
    for (int j = 0; j < 3; ++j) {
      const int hc = wave * 48 + j * 16;
      v8f gi = {}, gf = {}, gg = {}, go = {};
      for (int kc = 0; kc < Hh; kc += 32) {
        v16bf av = ld_k32(&Hbf[lane & 15][0], kc, hi);   // LDS A-operand
        gi = wmma_bf16(av, ld_k32(Whh + (size_t)(0 * Hh + hc + n) * Hh, kc, hi), gi);
        gf = wmma_bf16(av, ld_k32(Whh + (size_t)(1 * Hh + hc + n) * Hh, kc, hi), gf);
        gg = wmma_bf16(av, ld_k32(Whh + (size_t)(2 * Hh + hc + n) * Hh, kc, hi), gg);
        go = wmma_bf16(av, ld_k32(Whh + (size_t)(3 * Hh + hc + n) * Hh, kc, hi), go);
      }
      if (hi == 0) {  // hi==0 lanes cover C rows m=0..7 == all batch rows
#pragma unroll
        for (int r = 0; r < 8; ++r) {
          const int m = r, col = hc + n;
          const float* gx = Gx + ((size_t)m * steps + t) * H4;
          const float iv = gi[r] + gx[col]          + bvec[col];
          const float fv = gf[r] + gx[Hh + col]     + bvec[Hh + col];
          const float gv = gg[r] + gx[2 * Hh + col] + bvec[2 * Hh + col];
          const float ov = go[r] + gx[3 * Hh + col] + bvec[3 * Hh + col];
          const float si = 1.0f / (1.0f + __expf(-iv));
          const float sf = 1.0f / (1.0f + __expf(-fv));
          const float so = 1.0f / (1.0f + __expf(-ov));
          const float c2 = sf * cst[j][r] + si * tanhf(gv);
          const float h2 = so * tanhf(c2);
          cst[j][r] = c2;
          h2l[j][r] = h2;
          const size_t o = ((size_t)m * steps + t) * Hh + col;
          hs[o] = h2;
          if (hs_bf) hs_bf[o] = (bf16)h2;
          if (t == steps - 1) {
            if (hT) hT[m * Hh + col] = h2;
            if (cT) cT[m * Hh + col] = c2;
          }
        }
      }
    }
    __syncthreads();            // all waves done reading Hbf for step t
    if (hi == 0) {
#pragma unroll
      for (int j = 0; j < 3; ++j)
#pragma unroll
        for (int r = 0; r < 8; ++r)
          Hbf[r][wave * 48 + j * 16 + n] = (bf16)h2l[j][r];
    }
    __syncthreads();            // new h visible before next step
  }
}

// ---------------------------------------------------------------------------
// Small helper kernels
// ---------------------------------------------------------------------------
__global__ void cast_f32_bf16(bf16* __restrict__ dst, const float* __restrict__ src,
                              size_t nelem) {
  size_t i = (size_t)blockIdx.x * blockDim.x + threadIdx.x;
  const size_t stride = (size_t)gridDim.x * blockDim.x;
  for (; i < nelem; i += stride) dst[i] = (bf16)src[i];
}

__global__ void gather_embed_bf16(bf16* __restrict__ X, const int* __restrict__ ids,
                                  const float* __restrict__ emb) {
  const int tok = blockIdx.x;
  const float* s = emb + (size_t)ids[tok] * Ee;
  bf16* d = X + (size_t)tok * Ee;
  for (int i = threadIdx.x; i < Ee; i += blockDim.x) d[i] = (bf16)s[i];
}

__global__ void counters_kernel(int* __restrict__ ki_bt, int* __restrict__ an_bt,
                                const int* __restrict__ input_edits,
                                const int* __restrict__ org_ids) {
  const int b = threadIdx.x;
  if (b >= Bb) return;
  int kd = 0, ki = 0, an = 0;
  for (int t = 0; t < Tt - 1; ++t) {
    const int a = input_edits[b * Tt + t + 1];
    ki_bt[b * (Tt - 1) + t] = ki;   // pre-update values (matches the scan)
    an_bt[b * (Tt - 1) + t] = an;
    const int kd2 = kd + ((a == KEEP_ID) || (a == DEL_ID) ? 1 : 0);
    const bool not_special = (a != DEL_ID) && (a != STOP_ID) && (a != PAD_ID);
    const int ki2 = ki + (not_special ? 1 : 0);
    const bool is_ins = not_special && (a != KEEP_ID);
    const bool next_anno = org_ids[b * Ss + an + 1] == ANNO_ID;
    const int mx = kd2 > an ? kd2 : an;
    const int an2 = (is_ins && next_anno) ? (an + 1) : mx;
    kd = kd2; ki = ki2; an = an2;
  }
}

__global__ __launch_bounds__(256) void softmax_rows(float* __restrict__ x, int len) {
  float* p = x + (size_t)blockIdx.x * len;
  __shared__ float red[8];
  const int tid = threadIdx.x, lane = tid & 31, w = tid >> 5;
  float m = -INFINITY;
  for (int i = tid; i < len; i += 256) m = fmaxf(m, p[i]);
#pragma unroll
  for (int o = 16; o; o >>= 1) m = fmaxf(m, __shfl_xor(m, o, 32));
  if (!lane) red[w] = m;
  __syncthreads();
  float mm = red[0];
#pragma unroll
  for (int i = 1; i < 8; ++i) mm = fmaxf(mm, red[i]);
  __syncthreads();
  float s = 0.0f;
  for (int i = tid; i < len; i += 256) s += expf(p[i] - mm);
#pragma unroll
  for (int o = 16; o; o >>= 1) s += __shfl_xor(s, o, 32);
  if (!lane) red[w] = s;
  __syncthreads();
  float ss = 0.0f;
#pragma unroll
  for (int i = 0; i < 8; ++i) ss += red[i];
  const float inv = 1.0f / ss;
  for (int i = tid; i < len; i += 256) p[i] = expf(p[i] - mm) * inv;
}

__global__ __launch_bounds__(256) void log_softmax_rows(float* __restrict__ x, int len) {
  float* p = x + (size_t)blockIdx.x * len;
  __shared__ float red[8];
  const int tid = threadIdx.x, lane = tid & 31, w = tid >> 5;
  float m = -INFINITY;
  for (int i = tid; i < len; i += 256) m = fmaxf(m, p[i]);
#pragma unroll
  for (int o = 16; o; o >>= 1) m = fmaxf(m, __shfl_xor(m, o, 32));
  if (!lane) red[w] = m;
  __syncthreads();
  float mm = red[0];
#pragma unroll
  for (int i = 1; i < 8; ++i) mm = fmaxf(mm, red[i]);
  __syncthreads();
  float s = 0.0f;
  for (int i = tid; i < len; i += 256) s += expf(p[i] - mm);
#pragma unroll
  for (int o = 16; o; o >>= 1) s += __shfl_xor(s, o, 32);
  if (!lane) red[w] = s;
  __syncthreads();
  float ss = 0.0f;
#pragma unroll
  for (int i = 0; i < 8; ++i) ss += red[i];
  const float lz = mm + logf(ss);
  for (int i = tid; i < len; i += 256) p[i] -= lz;
}

__global__ __launch_bounds__(256) void ctx_kernel(float* __restrict__ ctx,
                                                  const float* __restrict__ attn,
                                                  const float* __restrict__ enc) {
  const int bt = blockIdx.x;                // b*T + t
  const int b = bt / Tt;
  __shared__ float arow[Ss];
  const float* ap = attn + (size_t)bt * Ss;
  for (int i = threadIdx.x; i < Ss; i += 256) arow[i] = ap[i];
  __syncthreads();
  const float* eb = enc + (size_t)b * Ss * Hh;
  for (int h = threadIdx.x; h < Hh; h += 256) {
    float acc = 0.0f;
    for (int s = 0; s < Ss; ++s) acc += arow[s] * eb[(size_t)s * Hh + h];
    ctx[(size_t)bt * Hh + h] = acc;
  }
}

__global__ void feat_kernel(bf16* __restrict__ feat, const float* __restrict__ hs_e,
                            const float* __restrict__ ctx, const float* __restrict__ enc,
                            const float* __restrict__ hs_w, const int* __restrict__ ki_bt,
                            const int* __restrict__ an_bt) {
  const int bt = blockIdx.x;                // b*(T-1) + t
  const int b = bt / (Tt - 1), t = bt % (Tt - 1);
  bf16* f = feat + (size_t)bt * H4;
  const float* oe = hs_e + ((size_t)b * Tt + t) * Hh;
  const float* cx = ctx + ((size_t)b * Tt + t) * Hh;
  const float* ce = enc + ((size_t)b * Ss + an_bt[bt]) * Hh;
  const float* cw = hs_w + ((size_t)b * Ss + ki_bt[bt]) * Hh;
  for (int i = threadIdx.x; i < Hh; i += blockDim.x) {
    f[i]          = (bf16)oe[i];
    f[Hh + i]     = (bf16)cx[i];
    f[2 * Hh + i] = (bf16)ce[i];
    f[3 * Hh + i] = (bf16)cw[i];
  }
}

// ---------------------------------------------------------------------------
// Host orchestration
// ---------------------------------------------------------------------------
extern "C" void kernel_launch(void* const* d_in, const int* in_sizes, int n_in,
                              void* d_out, int out_size, void* d_ws, size_t ws_size,
                              hipStream_t stream) {
  (void)in_sizes; (void)n_in; (void)out_size; (void)ws_size;

  const int*   input_edits = (const int*)d_in[0];
  const int*   org_ids     = (const int*)d_in[1];
  const int*   simp_sent   = (const int*)d_in[2];
  const float* h0     = (const float*)d_in[3];
  const float* c0     = (const float*)d_in[4];
  const float* Eo     = (const float*)d_in[5];
  const float* emb    = (const float*)d_in[6];
  const float* Wih_e  = (const float*)d_in[7];
  const float* Whh_e  = (const float*)d_in[8];
  const float* b_e    = (const float*)d_in[9];
  const float* Wih_w  = (const float*)d_in[10];
  const float* Whh_w  = (const float*)d_in[11];
  const float* b_w    = (const float*)d_in[12];
  const float* W_attn = (const float*)d_in[13];
  const float* W_align= (const float*)d_in[14];
  const float* W_mlp  = (const float*)d_in[15];
  const float* b_mlp  = (const float*)d_in[16];
  const float* W_out  = (const float*)d_in[17];
  const float* b_out  = (const float*)d_in[18];

  float* logp = (float*)d_out;                              // [B*(T-1), V]
  float* hT   = logp + (size_t)Bb * (Tt - 1) * Vv;          // [B, H]
  float* cT   = hT + (size_t)Bb * Hh;                       // [B, H]

  size_t off = 0;
  auto alloc = [&](size_t bytes) -> void* {
    void* p = (char*)d_ws + off;
    off += (bytes + 255) & ~(size_t)255;
    return p;
  };
  const size_t BT = (size_t)Bb * Tt, BS = (size_t)Bb * Ss, BTm1 = (size_t)Bb * (Tt - 1);

  bf16* WihE_b  = (bf16*)alloc(2 * (size_t)H4 * Ee);
  bf16* WhhE_b  = (bf16*)alloc(2 * (size_t)H4 * Hh);
  bf16* WihW_b  = (bf16*)alloc(2 * (size_t)H4 * Ee);
  bf16* WhhW_b  = (bf16*)alloc(2 * (size_t)H4 * Hh);
  bf16* Wattn_b = (bf16*)alloc(2 * (size_t)Hh * Hh);
  bf16* Walign_b= (bf16*)alloc(2 * (size_t)Hh * Ee);
  bf16* Wmlp_b  = (bf16*)alloc(2 * (size_t)Ee * H4);
  bf16* Wout_b  = (bf16*)alloc(2 * (size_t)Vv * Ee);
  bf16* Eo_b    = (bf16*)alloc(2 * BS * Ee);
  bf16* Xe_b    = (bf16*)alloc(2 * BT * Ee);
  bf16* Xw_b    = (bf16*)alloc(2 * BS * Ee);
  float* GxE    = (float*)alloc(4 * BT * H4);
  float* GxW    = (float*)alloc(4 * BS * H4);
  float* encf   = (float*)alloc(4 * BS * Hh);
  bf16* enc_b   = (bf16*)alloc(2 * BS * Hh);
  float* hsE    = (float*)alloc(4 * BT * Hh);
  bf16* hsE_b   = (bf16*)alloc(2 * BT * Hh);
  float* hsW    = (float*)alloc(4 * BS * Hh);
  float* keyf   = (float*)alloc(4 * BT * Hh);
  bf16* key_b   = (bf16*)alloc(2 * BT * Hh);
  float* scores = (float*)alloc(4 * BT * Ss);
  float* ctxf   = (float*)alloc(4 * BT * Hh);
  int*  ki_bt   = (int*)alloc(4 * BTm1);
  int*  an_bt   = (int*)alloc(4 * BTm1);
  bf16* feat_b  = (bf16*)alloc(2 * BTm1 * H4);
  float* mlpf   = (float*)alloc(4 * BTm1 * Ee);
  bf16* mlp_b   = (bf16*)alloc(2 * BTm1 * Ee);

  auto cast = [&](bf16* dst, const float* src, size_t nelem) {
    int blocks = (int)((nelem + 2047) / 2048);
    if (blocks > 4096) blocks = 4096;
    cast_f32_bf16<<<dim3(blocks), dim3(256), 0, stream>>>(dst, src, nelem);
  };
  // big GEMMs: async-LDS double-buffered block tiles
  auto gemmA = [&](float* C, bf16* Cbf, const bf16* A, const bf16* W,
                   const float* bias, int M, int N, int K, int ldc, int act) {
    dim3 g((N + 127) / 128, (M + 63) / 64);
    gemm_nt_bf16_async<<<g, dim3(256), 0, stream>>>(C, Cbf, A, W, bias, M, N, K,
                                                    ldc, act);
  };
  // batched direct GEMM (attention scores)
  auto gemmB = [&](float* C, bf16* Cbf, const bf16* A, const bf16* W,
                   const float* bias, int M, int N, int K, int ldc, int act,
                   int batch, long long sA, long long sW, long long sC) {
    const int mt = (M + 15) / 16, ng = (N + 63) / 64;
    const int bx = (mt * ng + 7) / 8;
    gemm_nt_bf16<<<dim3(bx, batch), dim3(256), 0, stream>>>(
        C, Cbf, A, W, bias, M, N, K, ldc, act, sA, sW, sC);
  };

  // 1) weight / input casts to bf16
  cast(WihE_b,  Wih_e,  (size_t)H4 * Ee);
  cast(WhhE_b,  Whh_e,  (size_t)H4 * Hh);
  cast(WihW_b,  Wih_w,  (size_t)H4 * Ee);
  cast(WhhW_b,  Whh_w,  (size_t)H4 * Hh);
  cast(Wattn_b, W_attn, (size_t)Hh * Hh);
  cast(Walign_b,W_align,(size_t)Hh * Ee);
  cast(Wmlp_b,  W_mlp,  (size_t)Ee * H4);
  cast(Wout_b,  W_out,  (size_t)Vv * Ee);
  cast(Eo_b,    Eo,     BS * Ee);

  // 2) embedding gathers
  gather_embed_bf16<<<dim3((int)BT), dim3(256), 0, stream>>>(Xe_b, input_edits, emb);
  gather_embed_bf16<<<dim3((int)BS), dim3(256), 0, stream>>>(Xw_b, simp_sent, emb);

  // 3) time-batched input projections + aligned encoder (tanh)
  gemmA(GxE, nullptr, Xe_b, WihE_b, nullptr, (int)BT, H4, Ee, H4, 0);
  gemmA(GxW, nullptr, Xw_b, WihW_b, nullptr, (int)BS, H4, Ee, H4, 0);
  gemmA(encf, enc_b, Eo_b, Walign_b, nullptr, (int)BS, Hh, Ee, Hh, 1);

  // 4) sequential LSTMs (persistent workgroup each)
  lstm_seq_kernel<<<dim3(1), dim3(512), 0, stream>>>(
      hsE, hsE_b, hT, cT, GxE, WhhE_b, b_e, h0, c0, Tt);
  lstm_seq_kernel<<<dim3(1), dim3(512), 0, stream>>>(
      hsW, nullptr, nullptr, nullptr, GxW, WhhW_b, b_w, h0, c0, Ss);

  // 5) counter recurrence
  counters_kernel<<<dim3(1), dim3(32), 0, stream>>>(ki_bt, an_bt, input_edits, org_ids);

  // 6) attention: key, scores (batched over B), softmax, ctx
  gemmA(keyf, key_b, hsE_b, Wattn_b, nullptr, (int)BT, Hh, Hh, Hh, 0);
  gemmB(scores, nullptr, key_b, enc_b, nullptr, Tt, Ss, Hh, Ss, 0, Bb,
        (long long)Tt * Hh, (long long)Ss * Hh, (long long)Tt * Ss);
  softmax_rows<<<dim3((int)BT), dim3(256), 0, stream>>>(scores, Ss);
  ctx_kernel<<<dim3((int)BT), dim3(256), 0, stream>>>(ctxf, scores, encf);

  // 7) feature concat (gathers via counters) -> mlp (tanh) -> logits -> logsoftmax
  feat_kernel<<<dim3((int)BTm1), dim3(256), 0, stream>>>(feat_b, hsE, ctxf, encf, hsW,
                                                         ki_bt, an_bt);
  gemmA(mlpf, mlp_b, feat_b, Wmlp_b, b_mlp, (int)BTm1, Ee, H4, Ee, 1);
  gemmA(logp, nullptr, mlp_b, Wout_b, b_out, (int)BTm1, Vv, Ee, Vv, 0);
  log_softmax_rows<<<dim3((int)BTm1), dim3(256), 0, stream>>>(logp, Vv);
}
